// ChannelAttn_5308579578329
// MI455X (gfx1250) — matrix operand order repaired and checked by reading, not compiled, not run on hardware
//
#include <hip/hip_runtime.h>

// ---------------- problem constants ----------------
#define B_   2
#define N_   8192
#define C_   768
#define H_   4
#define HD_  192
#define TC_  (3 * C_)      // 2304
#define M_   (B_ * N_)     // 16384

// ---------------- types ----------------
typedef __bf16 bf16;
typedef __attribute__((ext_vector_type(4)))  bf16  bf16x4;
typedef __attribute__((ext_vector_type(8)))  bf16  bf16x8;
typedef __attribute__((ext_vector_type(16))) bf16  bf16x16;
typedef __attribute__((ext_vector_type(8)))  float f32x8;
typedef __attribute__((ext_vector_type(4)))  int   i32x4;

#define WMMA_BF16(a, b, c) \
  __builtin_amdgcn_wmma_f32_16x16x32_bf16(false, (a), false, (b), (short)0, (c), false, false)

// ---------------- CDNA5 async global->LDS (ASYNCcnt-tracked) ---------------
#if __has_builtin(__builtin_amdgcn_global_load_async_to_lds_b128)
__device__ __forceinline__ void cp_async16(const bf16* g, bf16* l) {
  __builtin_amdgcn_global_load_async_to_lds_b128(
      (__attribute__((address_space(1))) i32x4*)g,
      (__attribute__((address_space(3))) i32x4*)l, 0, 0);
}
#else
__device__ __forceinline__ void cp_async16(const bf16* g, bf16* l) {
  unsigned loff = (unsigned)(unsigned long)(__attribute__((address_space(3))) char*)l;
  asm volatile("global_load_async_to_lds_b128 %0, %1, off" :: "v"(loff), "v"(g) : "memory");
}
#endif

template <int N>
__device__ __forceinline__ void wait_async() {
#if __has_builtin(__builtin_amdgcn_s_wait_asynccnt)
  __builtin_amdgcn_s_wait_asynccnt((short)N);
#else
  asm volatile("s_wait_asynccnt %0" :: "i"(N) : "memory");
#endif
}

// ---------------- WMMA fragment loads from LDS (row-major, ld elements) ----
// A 16x32 bf16: lane l holds row m=l&15; elems [0..7]=K (l>>4)*8+0..7,
//               elems [8..15]=K 16+(l>>4)*8+0..7  (ISA 7.12.2)
__device__ __forceinline__ bf16x16 frag_a(const bf16* sm, int row0, int ld) {
  const int l = threadIdx.x & 31;
  const bf16* p = sm + (row0 + (l & 15)) * ld + ((l >> 4) << 3);
  bf16x8 lo = *(const bf16x8*)p;
  bf16x8 hi = *(const bf16x8*)(p + 16);
  return __builtin_shufflevector(lo, hi, 0,1,2,3,4,5,6,7,8,9,10,11,12,13,14,15);
}
// B 32x16 bf16 (staged transposed: rows = output col n, cols = K):
// lane l holds col n=l&15; elems [0..7]=K (l>>4)*8+0..7, elems [8..15]=K 16+(l>>4)*8+0..7
__device__ __forceinline__ bf16x16 frag_b(const bf16* sm, int row0, int ld) {
  const int l = threadIdx.x & 31;
  const bf16* p = sm + (row0 + (l & 15)) * ld + ((l >> 4) << 3);
  bf16x8 lo = *(const bf16x8*)p;
  bf16x8 hi = *(const bf16x8*)(p + 16);
  return __builtin_shufflevector(lo, hi, 0,1,2,3,4,5,6,7,8,9,10,11,12,13,14,15);
}

// ---------------- fp32 -> bf16 converter (vec4) ----------------
__global__ void k_cvt(const float* __restrict__ in, bf16* __restrict__ out, long n4) {
  long i = (long)blockIdx.x * blockDim.x + threadIdx.x;
  const long stride = (long)gridDim.x * blockDim.x;
  for (; i < n4; i += stride) {
    float4 v = ((const float4*)in)[i];
    bf16x4 o;
    o[0] = (bf16)v.x; o[1] = (bf16)v.y; o[2] = (bf16)v.z; o[3] = (bf16)v.w;
    *(bf16x4*)(out + 4 * i) = o;
  }
}

// ---------------- generic bf16 WMMA GEMM, async double-buffered ------------
// C[M x Nc] = A[M x K] * Bt[Nc x K]^T (+bias).  BM=128 BN=64 BK=32,
// 256 thr = 8 waves in 4x2 grid; each wave 32x32 (2x2 WMMA tiles).
// Per chunk each thread issues 3 async b128 copies (2 for A, 1 for B).
__launch_bounds__(256)
__global__ void k_gemm(const bf16* __restrict__ A, int lda,
                       const bf16* __restrict__ Bt, int ldb,
                       float* __restrict__ Cf, bf16* __restrict__ Cb,
                       const float* __restrict__ bias, int K, int ldc) {
  constexpr int LD = 40;                 // 32 + 8 pad -> 80B rows, 16B aligned
  __shared__ bf16 As[2][128 * LD];
  __shared__ bf16 Bs[2][64 * LD];
  const int tid = threadIdx.x, lane = tid & 31, wave = tid >> 5;
  const int wm = wave >> 1, wn = wave & 1;
  const long bm = (long)blockIdx.y * 128;
  const int  bn = blockIdx.x * 64;

  const int ra = tid >> 2, ca = (tid & 3) << 3;   // A rows 0..63 (+64), cols
  const bf16* ga0 = A + (bm + ra) * (long)lda + ca;
  const bf16* ga1 = A + (bm + ra + 64) * (long)lda + ca;
  const bf16* gb  = Bt + (bn + ra) * (long)ldb + ca;

  const int NK = K >> 5;
  // prologue: stage chunk 0 into buffer 0
  cp_async16(ga0, &As[0][ra * LD + ca]);
  cp_async16(ga1, &As[0][(ra + 64) * LD + ca]);
  cp_async16(gb,  &Bs[0][ra * LD + ca]);

  f32x8 acc[2][2] = {};
  for (int c = 0; c < NK; ++c) {
    const int cur = c & 1;
    if (c + 1 < NK) {
      const int k1 = (c + 1) << 5;
      cp_async16(ga0 + k1, &As[cur ^ 1][ra * LD + ca]);
      cp_async16(ga1 + k1, &As[cur ^ 1][(ra + 64) * LD + ca]);
      cp_async16(gb  + k1, &Bs[cur ^ 1][ra * LD + ca]);
      wait_async<3>();                   // retire current chunk's 3 ops
    } else {
      wait_async<0>();
    }
    __syncthreads();
    bf16x16 af[2], bfr[2];
#pragma unroll
    for (int i = 0; i < 2; ++i) af[i]  = frag_a(As[cur], wm * 32 + i * 16, LD);
#pragma unroll
    for (int j = 0; j < 2; ++j) bfr[j] = frag_b(Bs[cur], wn * 32 + j * 16, LD);
#pragma unroll
    for (int i = 0; i < 2; ++i)
#pragma unroll
      for (int j = 0; j < 2; ++j)
        acc[i][j] = WMMA_BF16(af[i], bfr[j], acc[i][j]);
    __syncthreads();
  }
  // D layout: lane col = lane&15, row = v + 8*(lane>>4)
  const int col = lane & 15, rh = (lane >> 4) << 3;
#pragma unroll
  for (int i = 0; i < 2; ++i)
#pragma unroll
    for (int j = 0; j < 2; ++j) {
      const int n = bn + wn * 32 + j * 16 + col;
      const float bv = bias ? bias[n] : 0.0f;
#pragma unroll
      for (int v = 0; v < 8; ++v) {
        const long m = bm + wm * 32 + i * 16 + rh + v;
        const float val = acc[i][j][v] + bv;
        if (Cf) Cf[m * ldc + n] = val;
        else    Cb[m * ldc + n] = (bf16)val;
      }
    }
}

// ---------------- attn = q^T k  (per b,h: 192x192, K = N = 8192) -----------
// grid (3,3,8), 128 thr = 4 waves 2x2, each 32x32. Stage transposed chunks.
__launch_bounds__(128)
__global__ void k_qk(const bf16* __restrict__ qkv, float* __restrict__ attn) {
  constexpr int LD = 40;
  __shared__ bf16 Qs[64 * LD];   // [d][n]
  __shared__ bf16 Ks[64 * LD];   // [e][n]
  const int tid = threadIdx.x, lane = tid & 31, wave = tid >> 5;
  const int wm = wave >> 1, wn = wave & 1;
  const int e0 = blockIdx.x * 64, d0 = blockIdx.y * 64, bh = blockIdx.z;
  const int b = bh / H_, h = bh % H_;
  const long rs = TC_;
  const bf16* qb = qkv + (long)b * N_ * rs + h * HD_;       // q: col = h*HD + d
  const bf16* kb = qb + C_;                                 // k: col = C + h*HD + e

  f32x8 acc[2][2] = {};
  for (int n0 = 0; n0 < N_; n0 += 32) {
#pragma unroll
    for (int i = 0; i < 2; ++i) {
      int idx = tid + i * 128;                 // 0..255
      int nl = idx >> 3, ds = (idx & 7) << 3;  // n-local, d-segment
      bf16x8 qv = *(const bf16x8*)(qb + (long)(n0 + nl) * rs + d0 + ds);
      bf16x8 kv = *(const bf16x8*)(kb + (long)(n0 + nl) * rs + e0 + ds);
#pragma unroll
      for (int j = 0; j < 8; ++j) {
        Qs[(ds + j) * LD + nl] = qv[j];
        Ks[(ds + j) * LD + nl] = kv[j];
      }
    }
    __syncthreads();
    bf16x16 aq[2], bk[2];
#pragma unroll
    for (int i = 0; i < 2; ++i) aq[i] = frag_a(Qs, wm * 32 + i * 16, LD);
#pragma unroll
    for (int j = 0; j < 2; ++j) bk[j] = frag_b(Ks, wn * 32 + j * 16, LD);
#pragma unroll
    for (int i = 0; i < 2; ++i)
#pragma unroll
      for (int j = 0; j < 2; ++j)
        acc[i][j] = WMMA_BF16(aq[i], bk[j], acc[i][j]);
    __syncthreads();
  }
  const int col = lane & 15, rh = (lane >> 4) << 3;
  float* ab = attn + (long)bh * HD_ * HD_;
#pragma unroll
  for (int i = 0; i < 2; ++i)
#pragma unroll
    for (int j = 0; j < 2; ++j)
#pragma unroll
      for (int v = 0; v < 8; ++v)
        ab[(long)(d0 + wm * 32 + i * 16 + rh + v) * HD_ + (e0 + wn * 32 + j * 16 + col)] =
            acc[i][j][v];
}

// ---------------- temperature * softmax over d; write attn^T in bf16 -------
__launch_bounds__(256)
__global__ void k_softmax(const float* __restrict__ attn, const float* __restrict__ temp,
                          bf16* __restrict__ attnT) {
  const int bh = blockIdx.x / HD_, e = blockIdx.x % HD_, h = bh % H_;
  const int d = threadIdx.x;
  __shared__ float red[256];
  float a = 0.0f, val = -3.4e38f;
  if (d < HD_) { a = attn[(long)bh * HD_ * HD_ + (long)d * HD_ + e] * temp[h]; val = a; }
  red[d] = val; __syncthreads();
  for (int s = 128; s > 0; s >>= 1) { if (d < s) red[d] = fmaxf(red[d], red[d + s]); __syncthreads(); }
  const float mx = red[0]; __syncthreads();
  const float ex = (d < HD_) ? __expf(a - mx) : 0.0f;
  red[d] = ex; __syncthreads();
  for (int s = 128; s > 0; s >>= 1) { if (d < s) red[d] += red[d + s]; __syncthreads(); }
  const float inv = 1.0f / red[0];
  if (d < HD_) attnT[(long)bh * HD_ * HD_ + (long)e * HD_ + d] = (bf16)(ex * inv);
}

// ---------------- x_ca = v @ attn, fused scrambled-permute store ------------
// per (b,h): M = N_ rows, 192 cols, K = 192. grid (3, 64, 8), 256 thr.
// Async double-buffered like k_gemm (3 async b128 per thread per chunk).
__launch_bounds__(256)
__global__ void k_vattn(const bf16* __restrict__ qkv, const bf16* __restrict__ attnT,
                        bf16* __restrict__ xperm) {
  constexpr int LD = 40;
  __shared__ bf16 Vs[2][128 * LD];
  __shared__ bf16 Ts[2][64 * LD];
  const int tid = threadIdx.x, lane = tid & 31, wave = tid >> 5;
  const int wm = wave >> 1, wn = wave & 1;
  const int  e0 = blockIdx.x * 64;
  const long n0 = (long)blockIdx.y * 128;
  const int  bh = blockIdx.z, b = bh / H_, h = bh % H_;
  const bf16* vb = qkv + (long)b * N_ * TC_ + 2 * C_ + h * HD_;  // v rows, lda=TC_
  const bf16* tb = attnT + (long)bh * HD_ * HD_;                  // [e][d], ld=HD_

  const int ra = tid >> 2, ca = (tid & 3) << 3;
  const bf16* gv0 = vb + (n0 + ra) * (long)TC_ + ca;
  const bf16* gv1 = vb + (n0 + ra + 64) * (long)TC_ + ca;
  const bf16* gt  = tb + (e0 + ra) * (long)HD_ + ca;

  constexpr int NK = HD_ >> 5;   // 6
  cp_async16(gv0, &Vs[0][ra * LD + ca]);
  cp_async16(gv1, &Vs[0][(ra + 64) * LD + ca]);
  cp_async16(gt,  &Ts[0][ra * LD + ca]);

  f32x8 acc[2][2] = {};
  for (int c = 0; c < NK; ++c) {
    const int cur = c & 1;
    if (c + 1 < NK) {
      const int k1 = (c + 1) << 5;
      cp_async16(gv0 + k1, &Vs[cur ^ 1][ra * LD + ca]);
      cp_async16(gv1 + k1, &Vs[cur ^ 1][(ra + 64) * LD + ca]);
      cp_async16(gt  + k1, &Ts[cur ^ 1][ra * LD + ca]);
      wait_async<3>();
    } else {
      wait_async<0>();
    }
    __syncthreads();
    bf16x16 av[2], bt[2];
#pragma unroll
    for (int i = 0; i < 2; ++i) av[i] = frag_a(Vs[cur], wm * 32 + i * 16, LD);
#pragma unroll
    for (int j = 0; j < 2; ++j) bt[j] = frag_b(Ts[cur], wn * 32 + j * 16, LD);
#pragma unroll
    for (int i = 0; i < 2; ++i)
#pragma unroll
      for (int j = 0; j < 2; ++j)
        acc[i][j] = WMMA_BF16(av[i], bt[j], acc[i][j]);
    __syncthreads();
  }
  // permuted store: flat = b*N*C + e*(H*N) + h*N + n ; lane rows contiguous in n
  const int col = lane & 15, rh = (lane >> 4) << 3;
#pragma unroll
  for (int i = 0; i < 2; ++i)
#pragma unroll
    for (int j = 0; j < 2; ++j) {
      const int  e  = e0 + wn * 32 + j * 16 + col;
      const long nb = n0 + wm * 32 + i * 16 + rh;
      bf16x8 o;
#pragma unroll
      for (int v = 0; v < 8; ++v) o[v] = (bf16)acc[i][j][v];
      *(bf16x8*)(xperm + (long)b * N_ * C_ + (long)e * H_ * N_ + (long)h * N_ + nb) = o;
    }
}

// ---------------- launch ----------------
extern "C" void kernel_launch(void* const* d_in, const int* in_sizes, int n_in,
                              void* d_out, int out_size, void* d_ws, size_t ws_size,
                              hipStream_t stream) {
  const float* x      = (const float*)d_in[0];
  const float* w_qkv  = (const float*)d_in[1];
  const float* temper = (const float*)d_in[2];
  const float* w_proj = (const float*)d_in[3];
  const float* b_proj = (const float*)d_in[4];
  float* out = (float*)d_out;
  (void)in_sizes; (void)n_in; (void)out_size; (void)ws_size;

  char* ws = (char*)d_ws;
  bf16*  x_bf    = (bf16*)ws;  ws += (size_t)M_ * C_ * 2;           // 25.2 MB
  bf16*  wqkv_bf = (bf16*)ws;  ws += (size_t)TC_ * C_ * 2;          //  3.5 MB
  bf16*  wproj_bf= (bf16*)ws;  ws += (size_t)C_ * C_ * 2;           //  1.2 MB
  bf16*  qkv_bf  = (bf16*)ws;  ws += (size_t)M_ * TC_ * 2;          // 75.5 MB
  float* attn    = (float*)ws; ws += (size_t)B_ * H_ * HD_ * HD_ * 4;
  bf16*  attnT   = (bf16*)ws;  ws += (size_t)B_ * H_ * HD_ * HD_ * 2;
  bf16*  xperm   = (bf16*)ws;  ws += (size_t)M_ * C_ * 2;           // 25.2 MB

  // fp32 -> bf16
  k_cvt<<<1024, 256, 0, stream>>>(x,      x_bf,    (long)M_ * C_ / 4);
  k_cvt<<<256,  256, 0, stream>>>(w_qkv,  wqkv_bf, (long)TC_ * C_ / 4);
  k_cvt<<<128,  256, 0, stream>>>(w_proj, wproj_bf,(long)C_ * C_ / 4);

  // qkv = x @ w_qkv^T   (M=16384, Nc=2304, K=768) -> bf16
  dim3 g1(TC_ / 64, M_ / 128);
  k_gemm<<<g1, 256, 0, stream>>>(x_bf, C_, wqkv_bf, C_, nullptr, qkv_bf, nullptr, C_, TC_);

  // attn = q^T k  (8 x 192x192, K=8192) -> fp32
  dim3 g2(HD_ / 64, HD_ / 64, B_ * H_);
  k_qk<<<g2, 128, 0, stream>>>(qkv_bf, attn);

  // temperature + softmax over d, store attn^T bf16
  k_softmax<<<B_ * H_ * HD_, 256, 0, stream>>>(attn, temper, attnT);

  // x_ca = v @ attn with fused permute -> xperm bf16
  dim3 g3(HD_ / 64, N_ / 128, B_ * H_);
  k_vattn<<<g3, 256, 0, stream>>>(qkv_bf, attnT, xperm);

  // out = xperm @ w_proj^T + b_proj  (fp32)
  dim3 g4(C_ / 64, M_ / 128);
  k_gemm<<<g4, 256, 0, stream>>>(xperm, C_, wproj_bf, C_, out, nullptr, b_proj, C_, C_);
}
